// MultiHeadAttention_85607288143971
// MI455X (gfx1250) — matrix operand order repaired and checked by reading, not compiled, not run on hardware
//
#include <hip/hip_runtime.h>
#include <hip/hip_bf16.h>

// ---------------- problem constants ----------------
constexpr int CB = 2;      // batch
constexpr int CS = 2048;   // seq
constexpr int CE = 512;    // embed
constexpr int CH = 8;      // heads
constexpr int CD = 64;     // head dim
constexpr float CEPS = 1e-5f;

typedef __bf16 bf16_t;
typedef __attribute__((ext_vector_type(16))) __bf16 v16bf;
typedef __attribute__((ext_vector_type(8)))  float  v8f;

union ABf { v16bf v; unsigned u[8]; bf16_t h[16]; };
union CF  { v8f v; float f[8]; };

__device__ __forceinline__ v8f wmma_bf16(v16bf a, v16bf b, v8f c) {
  return __builtin_amdgcn_wmma_f32_16x16x32_bf16(false, a, false, b, (short)0, c, false, false);
}

// A operand, 16x32 bf16, element (m,k) at base[m*ld + k]. K-pairs contiguous -> 2x b128.
// Layout: lanes 0-15 row=lane, K 0..7 & 16..23; lanes 16-31 row=lane-16, K 8..15 & 24..31.
__device__ __forceinline__ v16bf load_A_rowmajor(const bf16_t* base, int ld, int lane) {
  const int row = lane & 15;
  const int kk  = (lane < 16) ? 0 : 8;
  const bf16_t* r = base + row * ld + kk;
  ABf a;
#pragma unroll
  for (int i = 0; i < 4; ++i) a.u[i]     = *(const unsigned*)(r + 2 * i);
#pragma unroll
  for (int i = 0; i < 4; ++i) a.u[4 + i] = *(const unsigned*)(r + 16 + 2 * i);
  return a.v;
}

// B operand, 32x16 bf16, element (k,n) at base[n*ldn + k] (K contiguous in memory -> 2x b128).
// Layout: lanes 0-15 col=lane, K 0..7 & 16..23; lanes 16-31 col=lane-16, K 8..15 & 24..31.
__device__ __forceinline__ v16bf load_B_kcontig(const bf16_t* base, int ldn, int lane) {
  const int n  = lane & 15;
  const int kk = (lane < 16) ? 0 : 8;
  const bf16_t* p = base + n * ldn + kk;
  ABf b;
#pragma unroll
  for (int i = 0; i < 8; ++i) b.u[i] = *(const unsigned*)(p + 2 * i + (i >= 4 ? 8 : 0));
  return b.v;
}

// ---------------- kernel 0a: fp32 -> bf16 cast ----------------
__global__ void cast_bf16_kernel(const float* __restrict__ src, bf16_t* __restrict__ dst, int n) {
  int i = blockIdx.x * 256 + threadIdx.x;
  if (i < n) dst[i] = (bf16_t)src[i];
}

// ---------------- kernel 0b: [H,E,D] fp32 -> [H,D,E] bf16 transpose-cast ----------------
// Makes the QKV weights K-contiguous when used as WMMA B operands.
__global__ void cast_transpose_hde_kernel(const float* __restrict__ src, bf16_t* __restrict__ dst) {
  int i = blockIdx.x * 256 + threadIdx.x;
  if (i < CH * CE * CD) {
    const int d = i % CD;
    const int e = (i / CD) % CE;
    const int h = i / (CD * CE);
    dst[((size_t)h * CD + d) * CE + e] = (bf16_t)src[i];
  }
}

// ---------------- kernel 1: QKV projection ----------------
// grid (S/16, H, B), block 32. q/k layout [B,H,S,D]; v written TRANSPOSED [B,H,D,S].
__global__ __launch_bounds__(32) void qkv_kernel(const bf16_t* __restrict__ xb,
                                                 const bf16_t* __restrict__ wqT,
                                                 const bf16_t* __restrict__ wkT,
                                                 const bf16_t* __restrict__ wvT,
                                                 bf16_t* __restrict__ q,
                                                 bf16_t* __restrict__ k,
                                                 bf16_t* __restrict__ vT) {
  const int lane = threadIdx.x & 31;
  const int m0 = blockIdx.x * 16;
  const int h  = blockIdx.y;
  const int b  = blockIdx.z;

  const bf16_t* xrow = xb + ((size_t)b * CS + m0) * CE;
  const bf16_t* wqh = wqT + (size_t)h * CD * CE;   // [D, E]
  const bf16_t* wkh = wkT + (size_t)h * CD * CE;
  const bf16_t* wvh = wvT + (size_t)h * CD * CE;

  CF accq[4], acck[4], accv[4];
#pragma unroll
  for (int nt = 0; nt < 4; ++nt)
#pragma unroll
    for (int r = 0; r < 8; ++r) { accq[nt].f[r] = 0.f; acck[nt].f[r] = 0.f; accv[nt].f[r] = 0.f; }

  for (int k0 = 0; k0 < CE; k0 += 32) {
    v16bf a = load_A_rowmajor(xrow + k0, CE, lane);
#pragma unroll
    for (int nt = 0; nt < 4; ++nt) {
      // B(k=e, n=d) = Wt[(n)*E + k] -> K contiguous
      v16bf bq = load_B_kcontig(wqh + (size_t)(nt * 16) * CE + k0, CE, lane);
      accq[nt].v = wmma_bf16(a, bq, accq[nt].v);
      v16bf bk = load_B_kcontig(wkh + (size_t)(nt * 16) * CE + k0, CE, lane);
      acck[nt].v = wmma_bf16(a, bk, acck[nt].v);
      v16bf bv = load_B_kcontig(wvh + (size_t)(nt * 16) * CE + k0, CE, lane);
      accv[nt].v = wmma_bf16(a, bv, accv[nt].v);
    }
  }

  // C layout: VGPR r -> row r + 8*(lane/16), col lane&15
  const size_t qkbase = ((size_t)(b * CH + h) * CS + m0) * CD;
  const size_t vbase  = (size_t)(b * CH + h) * CD * CS;   // [D, S]
  const int mo = (lane >> 4) * 8;
  const int nc = lane & 15;
#pragma unroll
  for (int nt = 0; nt < 4; ++nt)
#pragma unroll
    for (int r = 0; r < 8; ++r) {
      const int row = mo + r;            // seq position within tile
      const int col = nt * 16 + nc;      // d
      q[qkbase + (size_t)row * CD + col] = (bf16_t)accq[nt].f[r];
      k[qkbase + (size_t)row * CD + col] = (bf16_t)acck[nt].f[r];
      vT[vbase + (size_t)col * CS + (m0 + row)] = (bf16_t)accv[nt].f[r];
    }
}

// ---------------- kernel 2: causal flash attention ----------------
// grid (S/128, H, B), block 256 (8 waves, each owns 16 query rows).
// Writes vals[b, s, (H-1-h)*D + d] (the o[:, ::-1] head reversal folded into the store).
__global__ __launch_bounds__(256) void attn_kernel(const bf16_t* __restrict__ qb,
                                                   const bf16_t* __restrict__ kb,
                                                   const bf16_t* __restrict__ vbT,
                                                   bf16_t* __restrict__ vals) {
  __shared__ __align__(16) bf16_t pbuf[8][16 * 32];

  const int lane = threadIdx.x & 31;
  const int w    = threadIdx.x >> 5;
  const int h = blockIdx.y;
  const int b = blockIdx.z;
  const int m0 = blockIdx.x * 128 + w * 16;

  const bf16_t* qh  = qb  + (size_t)(b * CH + h) * CS * CD;   // [S, D]
  const bf16_t* kh  = kb  + (size_t)(b * CH + h) * CS * CD;   // [S, D]
  const bf16_t* vhT = vbT + (size_t)(b * CH + h) * CD * CS;   // [D, S]

  // Q tile (16 x 64) as two A operands
  v16bf aq0 = load_A_rowmajor(qh + (size_t)m0 * CD, CD, lane);
  v16bf aq1 = load_A_rowmajor(qh + (size_t)m0 * CD + 32, CD, lane);

  float rmax[8], rsum[8];
  CF acc[4];
#pragma unroll
  for (int r = 0; r < 8; ++r) { rmax[r] = -1e30f; rsum[r] = 0.f; }
#pragma unroll
  for (int nt = 0; nt < 4; ++nt)
#pragma unroll
    for (int r = 0; r < 8; ++r) acc[nt].f[r] = 0.f;

  const int mo = (lane >> 4) * 8;
  const int nc = lane & 15;
  bf16_t* pl = pbuf[w];

  const int tend = m0 + 16;
  for (int t0 = 0; t0 < tend; t0 += 32) {
    // warm the caches for the next key/value chunk (global_prefetch_b8)
    if (t0 + 32 < tend) {
      __builtin_prefetch(kh + (size_t)(t0 + 32) * CD + lane * 4, 0, 1);
      __builtin_prefetch(vhT + (size_t)lane * CS * 2 + (t0 + 32), 0, 1);
    }

    CF st[2];
#pragma unroll
    for (int tt = 0; tt < 2; ++tt) {
      const int tb = t0 + tt * 16;
      v8f s = {0.f, 0.f, 0.f, 0.f, 0.f, 0.f, 0.f, 0.f};
      // B(k=d, n=t) = k[(t)*D + d] -> K contiguous
      v16bf bk0 = load_B_kcontig(kh + (size_t)tb * CD, CD, lane);       // K = d 0..31
      s = wmma_bf16(aq0, bk0, s);
      v16bf bk1 = load_B_kcontig(kh + (size_t)tb * CD + 32, CD, lane);  // K = d 32..63
      s = wmma_bf16(aq1, bk1, s);
      st[tt].v = s;
      const int col = tb + nc;
#pragma unroll
      for (int r = 0; r < 8; ++r) {
        const int qr = m0 + mo + r;
        if (col > qr) st[tt].f[r] = -1e30f;   // causal mask
      }
    }

    // online softmax per row (rows live in 16-lane groups)
    float rowscale[8];
#pragma unroll
    for (int r = 0; r < 8; ++r) {
      float mx = fmaxf(st[0].f[r], st[1].f[r]);
      mx = fmaxf(mx, __shfl_xor(mx, 1, 32));
      mx = fmaxf(mx, __shfl_xor(mx, 2, 32));
      mx = fmaxf(mx, __shfl_xor(mx, 4, 32));
      mx = fmaxf(mx, __shfl_xor(mx, 8, 32));
      const float nm = fmaxf(rmax[r], mx);
      rowscale[r] = __expf(rmax[r] - nm);
      const float p0 = __expf(st[0].f[r] - nm);
      const float p1 = __expf(st[1].f[r] - nm);
      st[0].f[r] = p0; st[1].f[r] = p1;
      float ps = p0 + p1;
      ps += __shfl_xor(ps, 1, 32);
      ps += __shfl_xor(ps, 2, 32);
      ps += __shfl_xor(ps, 4, 32);
      ps += __shfl_xor(ps, 8, 32);
      rsum[r] = rsum[r] * rowscale[r] + ps;
      rmax[r] = nm;
    }
#pragma unroll
    for (int nt = 0; nt < 4; ++nt)
#pragma unroll
      for (int r = 0; r < 8; ++r) acc[nt].f[r] *= rowscale[r];

    // C-layout -> A-layout relayout of P through wave-private LDS
#pragma unroll
    for (int tt = 0; tt < 2; ++tt)
#pragma unroll
      for (int r = 0; r < 8; ++r)
        pl[(mo + r) * 32 + tt * 16 + nc] = (bf16_t)st[tt].f[r];

    v16bf pa = load_A_rowmajor(pl, 32, lane);
#pragma unroll
    for (int nt = 0; nt < 4; ++nt) {
      // B(k=t_local, n=d) = vT[(d)*S + t0 + t_local] -> K contiguous
      v16bf bv = load_B_kcontig(vhT + (size_t)(nt * 16) * CS + t0, CS, lane);
      acc[nt].v = wmma_bf16(pa, bv, acc[nt].v);
    }
  }

  // epilogue: normalize and store with head reversal
  const int hout = CH - 1 - h;
#pragma unroll
  for (int r = 0; r < 8; ++r) {
    const float inv = 1.f / rsum[r];
    const size_t rowbase = ((size_t)b * CS + (m0 + mo + r)) * CE + hout * CD;
#pragma unroll
    for (int nt = 0; nt < 4; ++nt)
      vals[rowbase + nt * 16 + nc] = (bf16_t)(acc[nt].f[r] * inv);
  }
}

// ---------------- kernel 3: out-proj + bias + residual + LayerNorm ----------------
// grid (B*S/16), block 256 (8 waves). Each block: 16 rows x 512 cols, K=512.
__global__ __launch_bounds__(256) void proj_ln_kernel(const bf16_t* __restrict__ valsb,
                                                      const bf16_t* __restrict__ wpb,
                                                      const float* __restrict__ x,
                                                      const float* __restrict__ bp,
                                                      const float* __restrict__ gamma,
                                                      const float* __restrict__ beta,
                                                      float* __restrict__ out) {
  __shared__ __align__(16) bf16_t astrip[16 * CE];
  __shared__ float redsum[16][8];
  __shared__ float redsq[16][8];

  const int lane = threadIdx.x & 31;
  const int w    = threadIdx.x >> 5;
  const int row0 = blockIdx.x * 16;

  // cooperative copy of the 16 x 512 bf16 A strip (contiguous) into LDS
  {
    const unsigned* src = (const unsigned*)(valsb + (size_t)row0 * CE);
    unsigned* dst = (unsigned*)astrip;
    for (int i = threadIdx.x; i < 16 * CE / 2; i += 256) dst[i] = src[i];
  }
  __syncthreads();

  const int n0 = w * 64;
  CF acc[4];
#pragma unroll
  for (int nt = 0; nt < 4; ++nt)
#pragma unroll
    for (int r = 0; r < 8; ++r) acc[nt].f[r] = 0.f;

  for (int k0 = 0; k0 < CE; k0 += 32) {
    // warm the caches for the next Wp K-slab
    if (k0 + 32 < CE)
      __builtin_prefetch(wpb + (size_t)(n0 + (lane & 15)) * CE + k0 + 32, 0, 1);

    v16bf a = load_A_rowmajor(astrip + k0, CE, lane);
#pragma unroll
    for (int nt = 0; nt < 4; ++nt) {
      // B(k=e, n=j) = Wp[j*E + e] -> K contiguous
      v16bf bb = load_B_kcontig(wpb + (size_t)(n0 + nt * 16) * CE + k0, CE, lane);
      acc[nt].v = wmma_bf16(a, bb, acc[nt].v);
    }
  }

  const int mo = (lane >> 4) * 8;
  const int nc = lane & 15;

  float vbuf[4][8];
  float psum[8], psq[8];
#pragma unroll
  for (int r = 0; r < 8; ++r) { psum[r] = 0.f; psq[r] = 0.f; }

#pragma unroll
  for (int nt = 0; nt < 4; ++nt) {
    const int j = n0 + nt * 16 + nc;
#pragma unroll
    for (int r = 0; r < 8; ++r) {
      const int row = row0 + mo + r;
      float val = acc[nt].f[r] + bp[j] + x[(size_t)row * CE + j];
      vbuf[nt][r] = val;
      psum[r] += val;
      psq[r]  += val * val;
    }
  }
#pragma unroll
  for (int r = 0; r < 8; ++r) {
    psum[r] += __shfl_xor(psum[r], 1, 32);
    psum[r] += __shfl_xor(psum[r], 2, 32);
    psum[r] += __shfl_xor(psum[r], 4, 32);
    psum[r] += __shfl_xor(psum[r], 8, 32);
    psq[r]  += __shfl_xor(psq[r], 1, 32);
    psq[r]  += __shfl_xor(psq[r], 2, 32);
    psq[r]  += __shfl_xor(psq[r], 4, 32);
    psq[r]  += __shfl_xor(psq[r], 8, 32);
  }
  if (nc == 0) {
#pragma unroll
    for (int r = 0; r < 8; ++r) { redsum[mo + r][w] = psum[r]; redsq[mo + r][w] = psq[r]; }
  }
  __syncthreads();

#pragma unroll
  for (int r = 0; r < 8; ++r) {
    const int M = mo + r;
    float s = 0.f, s2 = 0.f;
#pragma unroll
    for (int w2 = 0; w2 < 8; ++w2) { s += redsum[M][w2]; s2 += redsq[M][w2]; }
    const float mean = s * (1.f / CE);
    const float var  = s2 * (1.f / CE) - mean * mean;
    const float inv  = rsqrtf(var + CEPS);
    const int row = row0 + M;
#pragma unroll
    for (int nt = 0; nt < 4; ++nt) {
      const int j = n0 + nt * 16 + nc;
      out[(size_t)row * CE + j] = (vbuf[nt][r] - mean) * inv * gamma[j] + beta[j];
    }
  }
}

// ---------------- launch ----------------
extern "C" void kernel_launch(void* const* d_in, const int* in_sizes, int n_in,
                              void* d_out, int out_size, void* d_ws, size_t ws_size,
                              hipStream_t stream) {
  const float* x     = (const float*)d_in[0];
  const float* Wq    = (const float*)d_in[1];
  const float* Wk    = (const float*)d_in[2];
  const float* Wv    = (const float*)d_in[3];
  const float* Wp    = (const float*)d_in[4];
  const float* bp    = (const float*)d_in[5];
  const float* gamma = (const float*)d_in[6];
  const float* beta  = (const float*)d_in[7];
  // d_in[8] = mask: causal, implemented directly
  float* out = (float*)d_out;

  constexpr int NX = CB * CS * CE;       // 2097152
  constexpr int NW = CH * CE * CD;       // 262144 (== E*E for Wp)

  char* ws = (char*)d_ws;
  bf16_t* xb   = (bf16_t*)(ws);                          // 4 MB
  bf16_t* wqT  = (bf16_t*)(ws + 4194304);                // 512 KB each, [H,D,E]
  bf16_t* wkT  = (bf16_t*)(ws + 4194304 + 524288);
  bf16_t* wvT  = (bf16_t*)(ws + 4194304 + 2 * 524288);
  bf16_t* wpb  = (bf16_t*)(ws + 4194304 + 3 * 524288);   // [E,E] plain cast (already K-contig)
  bf16_t* qb   = (bf16_t*)(ws + 4194304 + 4 * 524288);   // 4 MB each
  bf16_t* kb   = (bf16_t*)(ws + 2 * 4194304 + 4 * 524288);
  bf16_t* vbT  = (bf16_t*)(ws + 3 * 4194304 + 4 * 524288);  // [B,H,D,S]
  bf16_t* vals = (bf16_t*)(ws + 4 * 4194304 + 4 * 524288);

  cast_bf16_kernel<<<(NX + 255) / 256, 256, 0, stream>>>(x,  xb,  NX);
  cast_transpose_hde_kernel<<<(NW + 255) / 256, 256, 0, stream>>>(Wq, wqT);
  cast_transpose_hde_kernel<<<(NW + 255) / 256, 256, 0, stream>>>(Wk, wkT);
  cast_transpose_hde_kernel<<<(NW + 255) / 256, 256, 0, stream>>>(Wv, wvT);
  cast_bf16_kernel<<<(NW + 255) / 256, 256, 0, stream>>>(Wp, wpb, NW);

  qkv_kernel<<<dim3(CS / 16, CH, CB), 32, 0, stream>>>(xb, wqT, wkT, wvT, qb, kb, vbT);
  attn_kernel<<<dim3(CS / 128, CH, CB), 256, 0, stream>>>(qb, kb, vbT, vals);
  proj_ln_kernel<<<dim3(CB * CS / 16), 256, 0, stream>>>(vals, wpb, x, bp, gamma, beta, out);
}
